// CrossAttention_25718264168859
// MI455X (gfx1250) — matrix operand order repaired and checked by reading, not compiled, not run on hardware
//
#include <hip/hip_runtime.h>
#include <hip/hip_bf16.h>

// ---------------------------------------------------------------------------
// CrossAttention on MI455X (gfx1250): bf16 WMMA everywhere, f32 accumulate.
// Pipeline: [proj Q][proj K][proj V] -> [flash attention] -> [out proj + bias]
// Intermediates (Q/K/V/attn in bf16) live in d_ws (~40 MiB, L2-resident).
// Row-major 16B-aligned tile staging uses GLOBAL_LOAD_ASYNC_TO_LDS_B128
// (ASYNCcnt-tracked) when the toolchain exposes the builtin.
// ---------------------------------------------------------------------------

typedef __bf16 bf16;
typedef bf16  v16bf __attribute__((ext_vector_type(16)));
typedef bf16  v8bf  __attribute__((ext_vector_type(8)));
typedef bf16  v4bf  __attribute__((ext_vector_type(4)));
typedef float v8f   __attribute__((ext_vector_type(8)));
typedef int   v4i   __attribute__((vector_size(16)));   // matches builtin param

#define SQ    4096
#define SKV   1024
#define BATCH 4
#define INNER 512
#define HSZ   64
#define NH    8

#if defined(__gfx1250__) && __has_builtin(__builtin_amdgcn_global_load_async_to_lds_b128)
#define HAVE_ASYNC_LDS 1
#else
#define HAVE_ASYNC_LDS 0
#endif

#if HAVE_ASYNC_LDS
// generic LDS pointer -> AS3: flat LDS address low 32 bits ARE the LDS byte
// address (ISA 10.2 aperture table), so int-truncate + inttoptr is exact.
static __device__ __forceinline__ void async_b128(const void* gsrc, void* ldst) {
  __builtin_amdgcn_global_load_async_to_lds_b128(
      (__attribute__((address_space(1))) v4i*)(unsigned long long)gsrc,
      (__attribute__((address_space(3))) v4i*)(unsigned)(unsigned long long)ldst,
      0, 0);
}
static __device__ __forceinline__ void async_wait0() {
#if __has_builtin(__builtin_amdgcn_s_wait_asynccnt)
  __builtin_amdgcn_s_wait_asynccnt(0);
#else
  asm volatile("s_wait_asynccnt 0x0" ::: "memory");
#endif
}
#endif

// f32 -> bf16 (hardware convert path on gfx1250)
static __device__ __forceinline__ bf16 tobf(float x) { return (bf16)x; }

// Build A-operand (16x32 bf16, contiguous-K layout): lane = row (lane&15), lane-half
// selects K-lo/K-hi. Elements 0..15 = K(half*16 + 0..15).
static __device__ __forceinline__ v16bf mk_a(const bf16* lo, const bf16* hi) {
  v8bf a = *(const v8bf*)lo;   // 16B aligned
  v8bf b = *(const v8bf*)hi;   // 16B aligned
  return __builtin_shufflevector(a, b, 0,1,2,3,4,5,6,7,8,9,10,11,12,13,14,15);
}

static __device__ __forceinline__ v8f wmma_bf16(v16bf a, v16bf b, v8f c) {
  // (neg_a, A, neg_b, B, c_mod, C, reuse_a, reuse_b)
  return __builtin_amdgcn_wmma_f32_16x16x32_bf16(false, a, false, b, (short)0, c,
                                                 false, false);
}

// ---------------------------------------------------------------------------
// Tiled GEMM: C[M,N] = A[M,Kd] * W[Kd,N]  (A: f32 or bf16; W: f32)
// Block: 256 threads (8 waves), tile 128x128, K-chunk 32.
// Wave w computes a 32x64 sub-tile: 2 M-tiles x 4 N-tiles of 16x16 WMMA.
// ---------------------------------------------------------------------------
template<bool A_IS_BF16, bool OUT_FP32_BIAS>
__global__ __launch_bounds__(256)
void gemm_wmma(const void* __restrict__ Av, const float* __restrict__ W,
               const float* __restrict__ bias, void* __restrict__ outv,
               int M, int N, int Kd) {
  __shared__ __align__(32) bf16 As[128 * 32];   // row-major A chunk
  __shared__ __align__(32) bf16 Bt[128 * 32];   // B chunk transposed [n][k]
  const int tid  = threadIdx.x;
  const int wave = tid >> 5;
  const int lane = tid & 31;
  const int hl   = lane >> 4;     // lane half
  const int ln   = lane & 15;
  const int m0 = blockIdx.y * 128;
  const int n0 = blockIdx.x * 128;
  const int waveM = (wave & 3) * 32;
  const int waveN = (wave >> 2) * 64;

  v8f zero = {};
  v8f acc[2][4];
#pragma unroll
  for (int i = 0; i < 2; ++i)
#pragma unroll
    for (int j = 0; j < 4; ++j) acc[i][j] = zero;

  for (int k0 = 0; k0 < Kd; k0 += 32) {
    // ---- stage A chunk ----
    if (A_IS_BF16) {
      const bf16* A = (const bf16*)Av;
#if HAVE_ASYNC_LDS
#pragma unroll
      for (int i = tid; i < 128 * 4; i += 256) {   // 16B chunks
        int r = i >> 2, c = i & 3;
        async_b128(A + (size_t)(m0 + r) * Kd + k0 + c * 8, As + r * 32 + c * 8);
      }
#else
#pragma unroll
      for (int i = tid; i < 128 * 8; i += 256) {
        int r = i >> 3, c4 = i & 7;
        *(v4bf*)(As + r * 32 + c4 * 4) =
            *(const v4bf*)(A + (size_t)(m0 + r) * Kd + k0 + c4 * 4);
      }
#endif
    } else {
      const float* A = (const float*)Av;
#pragma unroll
      for (int i = tid; i < 128 * 32; i += 256) {
        int r = i >> 5, c = i & 31;
        As[r * 32 + c] = tobf(A[(size_t)(m0 + r) * Kd + k0 + c]);
      }
    }
    // ---- stage W chunk transposed (B-operand wants contiguous K per column) ----
#pragma unroll
    for (int i = tid; i < 32 * 128; i += 256) {
      int k = i >> 7, n = i & 127;
      Bt[n * 32 + k] = tobf(W[(size_t)(k0 + k) * N + n0 + n]);
    }
    if (k0 + 32 < Kd)  // prefetch next weight chunk -> global_prefetch path
      __builtin_prefetch(W + (size_t)(k0 + 32) * N + n0 + (tid & 127), 0, 0);
#if HAVE_ASYNC_LDS
    if (A_IS_BF16) async_wait0();
#endif
    __syncthreads();

    v16bf aop[2];
#pragma unroll
    for (int i = 0; i < 2; ++i) {
      const bf16* rowp = As + (waveM + i * 16 + ln) * 32;
      aop[i] = mk_a(rowp + hl * 16, rowp + hl * 16 + 8);
    }
#pragma unroll
    for (int j = 0; j < 4; ++j) {
      v16bf bop = *(const v16bf*)(Bt + (waveN + j * 16 + ln) * 32 + hl * 16);
#pragma unroll
      for (int i = 0; i < 2; ++i) acc[i][j] = wmma_bf16(aop[i], bop, acc[i][j]);
    }
    __syncthreads();
  }

  // ---- writeback (C layout: row = r + 8*half, col = lane&15) ----
#pragma unroll
  for (int i = 0; i < 2; ++i)
#pragma unroll
    for (int j = 0; j < 4; ++j)
#pragma unroll
      for (int r = 0; r < 8; ++r) {
        int row = m0 + waveM + i * 16 + r + 8 * hl;
        int col = n0 + waveN + j * 16 + ln;
        float v = acc[i][j][r];
        if (OUT_FP32_BIAS)
          ((float*)outv)[(size_t)row * N + col] = v + bias[col];
        else
          ((bf16*)outv)[(size_t)row * N + col] = tobf(v);
      }
}

// ---------------------------------------------------------------------------
// Flash attention core: one block = (batch b, head h, 128 query rows).
// 8 waves x 16 query rows each. Skv consumed in 32-key chunks.
// ---------------------------------------------------------------------------
__global__ __launch_bounds__(256)
void attn_wmma(const bf16* __restrict__ Qb, const bf16* __restrict__ Kb,
               const bf16* __restrict__ Vb, bf16* __restrict__ Ob) {
  __shared__ __align__(32) bf16 Qs[128 * 64];   // Q tile, row-major      16 KB
  __shared__ __align__(32) bf16 Ks[32 * 64];    // K chunk, row-major      4 KB
  __shared__ __align__(32) bf16 Vt[64 * 32];    // V chunk transposed      4 KB
  __shared__ __align__(32) bf16 Ps[8][16 * 32]; // per-wave P scratch      8 KB

  const int tid  = threadIdx.x;
  const int wave = tid >> 5;
  const int lane = tid & 31;
  const int hl   = lane >> 4;
  const int ln   = lane & 15;
  const int q0 = blockIdx.x * 128;
  const int h  = blockIdx.y;
  const int b  = blockIdx.z;
  const size_t qbase  = ((size_t)b * SQ + q0) * INNER + h * HSZ;
  const size_t kvbase = ((size_t)b * SKV) * INNER + h * HSZ;

  // stage Q tile (async DMA to LDS when available)
#if HAVE_ASYNC_LDS
#pragma unroll
  for (int i = tid; i < 128 * 8; i += 256) {   // 16B chunks: 8 per 64-half row
    int r = i >> 3, c = i & 7;
    async_b128(Qb + qbase + (size_t)r * INNER + c * 8, Qs + r * 64 + c * 8);
  }
#else
#pragma unroll
  for (int i = tid; i < 128 * 16; i += 256) {
    int r = i >> 4, c4 = i & 15;
    *(v4bf*)(Qs + r * 64 + c4 * 4) =
        *(const v4bf*)(Qb + qbase + (size_t)r * INNER + c4 * 4);
  }
#endif

  v8f zero = {};
  v8f o[4];
#pragma unroll
  for (int j = 0; j < 4; ++j) o[j] = zero;
  float mrow[8], lrow[8];
#pragma unroll
  for (int r = 0; r < 8; ++r) { mrow[r] = -3.0e38f; lrow[r] = 0.0f; }
  const float scale = 0.125f;  // 1/sqrt(64)

  for (int t0 = 0; t0 < SKV; t0 += 32) {
    // stage K chunk (row-major, async)
#if HAVE_ASYNC_LDS
    {
      int r = tid >> 3, c = tid & 7;   // 32 rows x 8 x16B chunks = 256
      async_b128(Kb + kvbase + (size_t)(t0 + r) * INNER + c * 8,
                 Ks + r * 64 + c * 8);
    }
#else
#pragma unroll
    for (int i = tid; i < 32 * 16; i += 256) {
      int r = i >> 4, c4 = i & 15;
      *(v4bf*)(Ks + r * 64 + c4 * 4) =
          *(const v4bf*)(Kb + kvbase + (size_t)(t0 + r) * INNER + c4 * 4);
    }
#endif
    // stage V chunk transposed: Vt[d][kv]  (manual: transpose on the way in)
#pragma unroll
    for (int i = tid; i < 64 * 32; i += 256) {
      int d = i & 63, kv = i >> 6;
      Vt[d * 32 + kv] = Vb[kvbase + (size_t)(t0 + kv) * INNER + d];
    }
#if HAVE_ASYNC_LDS
    async_wait0();   // each wave drains its ASYNCcnt before the barrier
#endif
    __syncthreads();

    // S = Q * K^T  (two 16x16 N-tiles, K-dim 64 = 2 WMMA steps)
    v8f s0 = zero, s1 = zero;
    const bf16* qrow = Qs + (wave * 16 + ln) * 64;
#pragma unroll
    for (int kc = 0; kc < 64; kc += 32) {
      v16bf a  = mk_a(qrow + kc + hl * 16, qrow + kc + hl * 16 + 8);
      v16bf b0 = *(const v16bf*)(Ks + ln * 64 + kc + hl * 16);
      v16bf b1 = *(const v16bf*)(Ks + (16 + ln) * 64 + kc + hl * 16);
      s0 = wmma_bf16(a, b0, s0);
      s1 = wmma_bf16(a, b1, s1);
    }

    // online softmax (row = r + 8*hl lives in one 16-lane half -> xor 1/2/4/8)
    float corr[8], p0[8], p1[8];
#pragma unroll
    for (int r = 0; r < 8; ++r) {
      float x0 = s0[r] * scale, x1 = s1[r] * scale;
      float mx = fmaxf(x0, x1);
      mx = fmaxf(mx, __shfl_xor(mx, 1));
      mx = fmaxf(mx, __shfl_xor(mx, 2));
      mx = fmaxf(mx, __shfl_xor(mx, 4));
      mx = fmaxf(mx, __shfl_xor(mx, 8));
      float mnew = fmaxf(mrow[r], mx);
      p0[r] = __expf(x0 - mnew);
      p1[r] = __expf(x1 - mnew);
      float ps = p0[r] + p1[r];
      ps += __shfl_xor(ps, 1);
      ps += __shfl_xor(ps, 2);
      ps += __shfl_xor(ps, 4);
      ps += __shfl_xor(ps, 8);
      corr[r] = __expf(mrow[r] - mnew);
      lrow[r] = lrow[r] * corr[r] + ps;
      mrow[r] = mnew;
    }
#pragma unroll
    for (int j = 0; j < 4; ++j)
#pragma unroll
      for (int r = 0; r < 8; ++r) o[j][r] *= corr[r];

    // P: C-layout -> row-major bf16 in per-wave LDS scratch
    bf16* pw = Ps[wave];
#pragma unroll
    for (int r = 0; r < 8; ++r) {
      pw[(r + 8 * hl) * 32 + ln]      = tobf(p0[r]);
      pw[(r + 8 * hl) * 32 + 16 + ln] = tobf(p1[r]);
    }
    asm volatile("s_wait_dscnt 0" ::: "memory");  // own-wave LDS write->read

    // O += P * V  (4 d-tiles, K-dim 32 = 1 WMMA each)
    const bf16* prow = pw + ln * 32;
    v16bf ap = mk_a(prow + hl * 16, prow + hl * 16 + 8);
#pragma unroll
    for (int j = 0; j < 4; ++j) {
      v16bf bv = *(const v16bf*)(Vt + (j * 16 + ln) * 32 + hl * 16);
      o[j] = wmma_bf16(ap, bv, o[j]);
    }
    __syncthreads();
  }

  // normalize and store attn (bf16, same [B*Sq, INNER] layout as Q)
#pragma unroll
  for (int r = 0; r < 8; ++r) {
    float inv = 1.0f / lrow[r];
    size_t row = qbase + (size_t)(wave * 16 + r + 8 * hl) * INNER;
#pragma unroll
    for (int j = 0; j < 4; ++j)
      Ob[row + j * 16 + ln] = tobf(o[j][r] * inv);
  }
}

// ---------------------------------------------------------------------------
extern "C" void kernel_launch(void* const* d_in, const int* in_sizes, int n_in,
                              void* d_out, int out_size, void* d_ws, size_t ws_size,
                              hipStream_t stream) {
  const float* inputs  = (const float*)d_in[0];  // [4,4096,512]
  const float* context = (const float*)d_in[1];  // [4,1024,768]
  const float* Wq = (const float*)d_in[2];       // [512,512]
  const float* Wk = (const float*)d_in[3];       // [768,512]
  const float* Wv = (const float*)d_in[4];       // [768,512]
  const float* Wo = (const float*)d_in[5];       // [512,512]
  const float* bo = (const float*)d_in[6];       // [512]
  float* out = (float*)d_out;

  char* ws = (char*)d_ws;
  bf16* Qb = (bf16*)(ws);                                  // 16 MiB
  bf16* Kb = (bf16*)(ws + (size_t)16 * 1024 * 1024);       //  4 MiB
  bf16* Vb = (bf16*)(ws + (size_t)20 * 1024 * 1024);       //  4 MiB
  bf16* Ab = (bf16*)(ws + (size_t)24 * 1024 * 1024);       // 16 MiB

  dim3 blk(256);
  // Q = inputs @ Wq   : M=16384 K=512 N=512
  gemm_wmma<false, false><<<dim3(4, 128), blk, 0, stream>>>(inputs, Wq, nullptr, Qb,
                                                            16384, 512, 512);
  // K = context @ Wk  : M=4096 K=768 N=512
  gemm_wmma<false, false><<<dim3(4, 32), blk, 0, stream>>>(context, Wk, nullptr, Kb,
                                                           4096, 512, 768);
  // V = context @ Wv
  gemm_wmma<false, false><<<dim3(4, 32), blk, 0, stream>>>(context, Wv, nullptr, Vb,
                                                           4096, 512, 768);
  // fused scaled-dot-product attention (flash style)
  attn_wmma<<<dim3(SQ / 128, NH, BATCH), blk, 0, stream>>>(Qb, Kb, Vb, Ab);
  // out = attn @ Wo + bo  (f32 output)
  gemm_wmma<true, true><<<dim3(4, 128), blk, 0, stream>>>(Ab, Wo, bo, out,
                                                          16384, 512, 512);
}
